// QuantLinear_82557861364247
// MI455X (gfx1250) — matrix operand order repaired and checked by reading, not compiled, not run on hardware
//
#include <hip/hip_runtime.h>

// ---------------------------------------------------------------------------
// W4A16 quantized GEMM for MI455X (gfx1250, wave32, WMMA f16->f32)
// out[B,T] = x[B,K] @ dequant(qweight[K/8,T], scales[G,T], zeros[G,T]) + bias
// Tile 128x128 (256 threads = 8 wave32), K-step 64, double-buffered LDS.
// Staging is all 16-bit packed math:
//   x:  v_cvt_pk_rtz_f16_f32 (1 op / 2 elems)
//   w4: magic 0x6400|n == f16(1024+n)  ->  pk_add(-1024) -> pk_fma(s, -z*s)
// ---------------------------------------------------------------------------

typedef __attribute__((ext_vector_type(16))) _Float16 v16h;
typedef __attribute__((ext_vector_type(8)))  _Float16 v8h;
typedef __attribute__((ext_vector_type(4)))  _Float16 v4h;
typedef __attribute__((ext_vector_type(2)))  _Float16 v2h;
typedef __attribute__((ext_vector_type(8)))  float    v8f;

#define BM 128
#define BN 128
#define BK 64
#define LDA 72   // BK + 8 pad; 144B pitch = 9*16B (keeps b128 alignment)
#define LDB 72

__device__ __forceinline__ v2h pk_f16(float a, float b) {
  // v_cvt_pk_rtz_f16_f32; builtin returns __fp16x2 — bit-identical to v2h
  return __builtin_bit_cast(v2h, __builtin_amdgcn_cvt_pkrtz(a, b));
}

__global__ __launch_bounds__(256)
void qgemm_w4a16_f16(const float* __restrict__ x,
                     const int*   __restrict__ qw,
                     const float* __restrict__ scales,
                     const float* __restrict__ zeros,
                     const float* __restrict__ bias,
                     float* __restrict__ out,
                     int K, int T) {
  __shared__ __align__(16) _Float16 As[2 * BM * LDA];
  __shared__ __align__(16) _Float16 Bs[2 * BN * LDB];

  const int tid    = threadIdx.x;
  const int lane   = tid & 31;
  const int wave   = tid >> 5;          // 0..7
  const int laneM  = lane & 15;
  const int laneHi = lane >> 4;
  const int wm     = (wave >> 2) * 64;  // wave M offset: 0 / 64
  const int wn     = (wave & 3) * 32;   // wave N offset: 0,32,64,96

  const int rowbase = blockIdx.y * BM;
  const int colbase = blockIdx.x * BN;

  // x tile: 128 rows x 64 cols f32. tid%16 = 4-col quad (16 threads cover one
  // row's 256B contiguously), tid/16 = base row; each thread covers 8 rows.
  const int kq  = tid & 15;
  const int r0x = tid >> 4;
  // weight tile: 64 k x 128 n -> 8 qweight rows x 128 cols int32.
  // tid%128 = column, tid/128 selects rows {h*4 .. h*4+3} (k = h*32 + 8j..).
  const int wcol  = tid & 127;
  const int whalf = tid >> 7;

  const int ksteps = K / BK;

  // staged global-load registers
  float4 xr[8];
  unsigned wq[4];
  float sS = 0.f, sC = 0.f;   // scale, -zero*scale

  v8f acc[4][2];
#pragma unroll
  for (int mt = 0; mt < 4; ++mt)
#pragma unroll
    for (int nt = 0; nt < 2; ++nt)
#pragma unroll
      for (int v = 0; v < 8; ++v) acc[mt][nt][v] = 0.f;

#define GLOAD(step)                                                            \
  {                                                                            \
    const int kb = (step) * BK;                                                \
    _Pragma("unroll")                                                          \
    for (int i = 0; i < 8; ++i)                                                \
      xr[i] = *(const float4*)(x + (size_t)(rowbase + r0x + 16 * i) * K + kb + \
                               kq * 4);                                        \
    const int krb = kb >> 3;                                                   \
    const int g   = kb >> 7; /* GROUP_SIZE=128, BK=64 -> one group per tile */ \
    const size_t col = (size_t)colbase + wcol;                                 \
    _Pragma("unroll")                                                          \
    for (int j = 0; j < 4; ++j)                                                \
      wq[j] = (unsigned)qw[(size_t)(krb + whalf * 4 + j) * T + col];           \
    const float s = scales[(size_t)g * T + col];                               \
    sS = s;                                                                    \
    sC = -zeros[(size_t)g * T + col] * s;                                      \
    if ((step) + 4 < ksteps) { /* gfx1250 global_prefetch_b8 */                \
      const int kbp = ((step) + 4) * BK;                                       \
      __builtin_prefetch(x + (size_t)(rowbase + r0x) * K + kbp + kq * 4, 0,    \
                         1);                                                   \
      __builtin_prefetch(qw + (size_t)((kbp >> 3) + whalf * 4) * T + col, 0,   \
                         1);                                                   \
    }                                                                          \
  }

#define LSTORE(bufi)                                                           \
  {                                                                            \
    _Float16* Ad = As + (bufi) * (BM * LDA);                                   \
    _Pragma("unroll")                                                          \
    for (int i = 0; i < 8; ++i) {                                              \
      v2h p0 = pk_f16(xr[i].x, xr[i].y);                                       \
      v2h p1 = pk_f16(xr[i].z, xr[i].w);                                       \
      v4h t  = __builtin_shufflevector(p0, p1, 0, 1, 2, 3);                    \
      *(v4h*)(Ad + (r0x + 16 * i) * LDA + kq * 4) = t;                         \
    }                                                                          \
    _Float16* Bd =                                                             \
        Bs + (bufi) * (BN * LDB) + wcol * LDB + whalf * 32;                    \
    const _Float16 sh = (_Float16)sS;                                          \
    const _Float16 ch = (_Float16)sC;                                          \
    const v2h s2 = {sh, sh};                                                   \
    const v2h c2 = {ch, ch};                                                   \
    const v2h kThou = {(_Float16)1024.0f, (_Float16)1024.0f};                  \
    _Pragma("unroll")                                                          \
    for (int j = 0; j < 4; ++j) {                                              \
      const unsigned wv = wq[j];                                               \
      v2h pk[4];                                                               \
      _Pragma("unroll")                                                        \
      for (int p = 0; p < 4; ++p) {                                            \
        const unsigned b = wv >> (8 * p);                                      \
        const unsigned e =                                                     \
            0x64006400u | (b & 0xFu) | ((b & 0xF0u) << 12);                    \
        v2h qh = __builtin_bit_cast(v2h, e);                                   \
        pk[p] = __builtin_elementwise_fma(qh - kThou, s2, c2);                 \
      }                                                                        \
      v4h lo = __builtin_shufflevector(pk[0], pk[1], 0, 1, 2, 3);              \
      v4h hi = __builtin_shufflevector(pk[2], pk[3], 0, 1, 2, 3);              \
      v8h t  = __builtin_shufflevector(lo, hi, 0, 1, 2, 3, 4, 5, 6, 7);        \
      *(v8h*)(Bd + j * 8) = t;                                                 \
    }                                                                          \
  }

  GLOAD(0);

  for (int s = 0; s < ksteps; ++s) {
    const int buf = s & 1;
    LSTORE(buf);
    __syncthreads();
    if (s + 1 < ksteps) GLOAD(s + 1);

    const _Float16* A0 = As + buf * (BM * LDA);
    const _Float16* B0 = Bs + buf * (BN * LDB);

#pragma unroll
    for (int kk = 0; kk < BK; kk += 32) {
      // A fragments (16x32 f16 wave32 layout: k-base=(lane/16)*8, chunks at
      // +0 / +16) — two ds_load_b128 per tile
      v16h af[4];
#pragma unroll
      for (int mt = 0; mt < 4; ++mt) {
        const _Float16* p =
            A0 + (wm + mt * 16 + laneM) * LDA + kk + laneHi * 8;
        v8h lo = *(const v8h*)(p);
        v8h hi = *(const v8h*)(p + 16);
        af[mt] = __builtin_shufflevector(lo, hi, 0, 1, 2, 3, 4, 5, 6, 7, 8, 9,
                                         10, 11, 12, 13, 14, 15);
      }
      // B fragments (32x16 f16: lane holds col n=lane%16, k-base=(lane/16)*8,
      // chunks at +0 / +16)
      v16h bf[2];
#pragma unroll
      for (int nt = 0; nt < 2; ++nt) {
        const _Float16* p =
            B0 + (wn + nt * 16 + laneM) * LDB + kk + laneHi * 8;
        v8h lo = *(const v8h*)(p);
        v8h hi = *(const v8h*)(p + 16);
        bf[nt] = __builtin_shufflevector(lo, hi, 0, 1, 2, 3, 4, 5, 6, 7, 8, 9,
                                         10, 11, 12, 13, 14, 15);
      }

#pragma unroll
      for (int mt = 0; mt < 4; ++mt)
#pragma unroll
        for (int nt = 0; nt < 2; ++nt)
          acc[mt][nt] = __builtin_amdgcn_wmma_f32_16x16x32_f16(
              /*neg_a=*/false, af[mt], /*neg_b=*/false, bf[nt],
              /*c_mod=*/(short)0, acc[mt][nt], /*reuse_a=*/false,
              /*reuse_b=*/false);
    }
  }

  // epilogue: C layout — VGPR v: m=(lane/16)*8+v, n=lane%16
#pragma unroll
  for (int nt = 0; nt < 2; ++nt) {
    const int col = colbase + wn + nt * 16 + laneM;
    const float bv = bias[col];
#pragma unroll
    for (int mt = 0; mt < 4; ++mt) {
      const int rbase = rowbase + wm + mt * 16 + laneHi * 8;
#pragma unroll
      for (int v = 0; v < 8; ++v)
        out[(size_t)(rbase + v) * T + col] = acc[mt][nt][v] + bv;
    }
  }
}

extern "C" void kernel_launch(void* const* d_in, const int* in_sizes, int n_in,
                              void* d_out, int out_size, void* d_ws,
                              size_t ws_size, hipStream_t stream) {
  const float* x      = (const float*)d_in[0];
  const int*   qw     = (const int*)d_in[1];
  const float* scales = (const float*)d_in[2];
  const float* zeros  = (const float*)d_in[3];
  const float* bias   = (const float*)d_in[4];
  float*       out    = (float*)d_out;

  const int T = in_sizes[4];            // 11008
  const int K = (in_sizes[1] / T) * 8;  // qweight rows = K/8
  const int B = in_sizes[0] / K;        // 4096

  dim3 grid(T / BN, B / BM);            // (86, 32)
  dim3 block(256);
  qgemm_w4a16_f16<<<grid, block, 0, stream>>>(x, qw, scales, zeros, bias, out,
                                              K, T);
}
